// MultiheadedAttention_11613591568744
// MI455X (gfx1250) — compile-verified, matrix operand order checked by reading, not run on hardware
//
#include <hip/hip_runtime.h>
#include <hip/hip_bf16.h>

#define BB 2
#define SS 4096
#define EE 512
#define HH 8
#define DD 64
#define HD 512
#define MM (BB * SS)

typedef __attribute__((ext_vector_type(16))) __bf16 v16bf;
typedef __attribute__((ext_vector_type(8)))  float  v8f;

// Native clang vector (NOT HIP_vector_type) for the async-to-LDS builtin.
typedef int gnu_v4i __attribute__((vector_size(16)));
typedef __attribute__((address_space(1))) gnu_v4i as1_v4i;
typedef __attribute__((address_space(3))) gnu_v4i as3_v4i;

union Frag { v16bf v; int4 q[2]; };

__device__ __forceinline__ v8f vzero8() {
  v8f z = {0.f, 0.f, 0.f, 0.f, 0.f, 0.f, 0.f, 0.f};
  return z;
}

__device__ __forceinline__ v8f wmma_bf16(const Frag& a, const Frag& b, v8f c) {
  return __builtin_amdgcn_wmma_f32_16x16x32_bf16(
      /*neg_a=*/false, a.v, /*neg_b=*/false, b.v,
      /*c_mod=*/(short)0, c, /*reuse_a=*/false, /*reuse_b=*/false);
}

// 16-byte global -> LDS copy. Prefer the CDNA5 async-to-LDS path (ASYNCcnt).
__device__ __forceinline__ void cp_async16(const void* g, void* l) {
#if defined(__gfx1250__) && __has_builtin(__builtin_amdgcn_global_load_async_to_lds_b128)
  __builtin_amdgcn_global_load_async_to_lds_b128((as1_v4i*)g, (as3_v4i*)l, 0, 0);
#else
  *(int4*)l = *(const int4*)g;
#endif
}

template <int N>
__device__ __forceinline__ void cp_wait() {
#if defined(__gfx1250__) && __has_builtin(__builtin_amdgcn_global_load_async_to_lds_b128)
#if __has_builtin(__builtin_amdgcn_s_wait_asynccnt)
  __builtin_amdgcn_s_wait_asynccnt(N);
#else
  asm volatile("s_wait_asynccnt %0" :: "i"(N) : "memory");
#endif
#endif
}

// ---------------------------------------------------------------------------
// Tiled GEMM: C[M x 512] = A[M x 512] * Bw[512 x 512], bf16 WMMA, f32 acc.
// Block tile 128(M) x 64(N), 8 waves, each wave 32x32 (2x2 WMMA).
// AMODE: 0 = A f32 (convert at staging), 1 = A bf16 (async-copy staging).
// EMODE: 0 = f32 out [M,512]; 1 = bf16 [B,H,S,D]; 2 = bf16 [B,H,D,S];
//        3 = bf16 [B,H,S,D] pre-scaled by 1/sqrt(D) (Q).
// ---------------------------------------------------------------------------
template <int AMODE, int EMODE>
__global__ __launch_bounds__(256) void gemm_wmma(const void* __restrict__ Aptr,
                                                 const float* __restrict__ Bw,
                                                 void* __restrict__ outp) {
  __shared__ __bf16 As[8][32][16];  // 8 M-subtiles, WMMA A-operand layout
  __shared__ __bf16 Bs[4][32][16];  // 4 N-subtiles, WMMA B-operand layout

  const int tid  = threadIdx.x;
  const int lane = tid & 31;
  const int w    = tid >> 5;
  const int wm   = w >> 1;
  const int wn   = w & 1;
  const int blockM = blockIdx.x * 128;
  const int blockN = blockIdx.y * 64;

  v8f acc[2][2];
#pragma unroll
  for (int i = 0; i < 2; ++i)
#pragma unroll
    for (int j = 0; j < 2; ++j) acc[i][j] = vzero8();

  const int sm = tid >> 1;  // A row 0..127
  const int sh = tid & 1;   // 16-wide K group
  const int bn = tid & 63;  // B col 0..63
  const int ba = tid >> 6;  // B k-group (8 consecutive k)

  for (int k0 = 0; k0 < 512; k0 += 32) {
    __syncthreads();
    // ---- stage A tile (128 x 32) ----
    {
      const int sub = sm >> 4, ml = sm & 15, joff = sh * 8;
      if (AMODE == 0) {
        const float* ap =
            (const float*)Aptr + (size_t)(blockM + sm) * 512 + k0 + sh * 16;
        __bf16 tmp[16];
#pragma unroll
        for (int i = 0; i < 16; ++i) tmp[i] = (__bf16)ap[i];
        *(int4*)&As[sub][ml][joff]      = *(int4*)&tmp[0];
        *(int4*)&As[sub][ml + 16][joff] = *(int4*)&tmp[8];
      } else {
        const __bf16* ap =
            (const __bf16*)Aptr + (size_t)(blockM + sm) * 512 + k0 + sh * 16;
        cp_async16(ap,     &As[sub][ml][joff]);
        cp_async16(ap + 8, &As[sub][ml + 16][joff]);
      }
    }
    // ---- stage B tile (32 x 64), f32 -> bf16 ----
    {
      const float* bp = Bw + (size_t)(k0 + ba * 8) * 512 + blockN + bn;
      __bf16 tb[8];
#pragma unroll
      for (int i = 0; i < 8; ++i) tb[i] = (__bf16)bp[(size_t)i * 512];
      const int h = ba & 1, joff = (ba >> 1) * 8;
      *(int4*)&Bs[bn >> 4][(bn & 15) + 16 * h][joff] = *(int4*)&tb[0];
    }
    if (AMODE == 1) cp_wait<0>();
    __syncthreads();

    Frag af[2], bf[2];
#pragma unroll
    for (int i = 0; i < 2; ++i) {
      af[i].q[0] = *(int4*)&As[2 * wm + i][lane][0];
      af[i].q[1] = *(int4*)&As[2 * wm + i][lane][8];
      bf[i].q[0] = *(int4*)&Bs[2 * wn + i][lane][0];
      bf[i].q[1] = *(int4*)&Bs[2 * wn + i][lane][8];
    }
#pragma unroll
    for (int i = 0; i < 2; ++i)
#pragma unroll
      for (int j = 0; j < 2; ++j) acc[i][j] = wmma_bf16(af[i], bf[j], acc[i][j]);
  }

  // ---- epilogue ----
#pragma unroll
  for (int i = 0; i < 2; ++i)
#pragma unroll
    for (int j = 0; j < 2; ++j)
#pragma unroll
      for (int r = 0; r < 8; ++r) {
        const int row = blockM + wm * 32 + i * 16 + r + 8 * (lane >> 4);
        const int col = blockN + wn * 32 + j * 16 + (lane & 15);
        float v = acc[i][j][r];
        if (EMODE == 0) {
          ((float*)outp)[(size_t)row * 512 + col] = v;
        } else {
          if (EMODE == 3) v *= 0.125f;  // fold 1/sqrt(D) into Q
          const int b = row >> 12, s = row & (SS - 1);
          const int hh = col >> 6, d = col & 63;
          __bf16* o = (__bf16*)outp;
          if (EMODE == 2)
            o[(((size_t)b * HH + hh) * DD + d) * SS + s] = (__bf16)v;  // [B,H,D,S]
          else
            o[(((size_t)b * HH + hh) * SS + s) * DD + d] = (__bf16)v;  // [B,H,S,D]
        }
      }
}

// ---------------------------------------------------------------------------
// Flash attention. Block = 8 waves sharing one (b,h); wave handles 16 q-rows.
// K/V/mask tiles for each 32-wide Sk block are staged cooperatively into LDS
// with double-buffered async global->LDS copies (3 async b128 per thread per
// block), eliminating the 8x per-wave redundancy of direct global fragment
// loads. Q is pre-scaled by 1/sqrt(D) at projection time.
// ---------------------------------------------------------------------------
__global__ __launch_bounds__(256) void flash_attn(
    const __bf16* __restrict__ Qh, const __bf16* __restrict__ Kh,
    const __bf16* __restrict__ Vt, const unsigned char* __restrict__ mask,
    __bf16* __restrict__ attnout) {
  __shared__ __bf16 Klds[2][32][64];          // [buf][sk][d]
  __shared__ __bf16 Vlds[2][64][32];          // [buf][d][sk]
  __shared__ unsigned char Mlds[2][128][32];  // [buf][q_local][sk]
  __shared__ __bf16 Plds[8][16][32];          // per-wave P C->A transpose

  const int tid  = threadIdx.x;
  const int lane = tid & 31;
  const int w    = tid >> 5;
  const int bh   = blockIdx.x >> 5;
  const int qc   = blockIdx.x & 31;
  const int q0b  = qc * 128;
  const int q0   = q0b + w * 16;
  const int mrow = lane & 15;
  const int hb   = lane >> 4;

  // Q A-fragments (pre-scaled), resident in registers
  Frag aq[2];
  {
    const __bf16* Qbase = Qh + ((size_t)bh * SS + q0 + mrow) * DD;
#pragma unroll
    for (int c = 0; c < 2; ++c) {
      const __bf16* qp = Qbase + c * 32 + 8 * hb;
      aq[c].q[0] = *(const int4*)qp;
      aq[c].q[1] = *(const int4*)(qp + 16);
    }
  }

  const __bf16* Kb = Kh + (size_t)bh * SS * DD;
  const __bf16* Vb = Vt + (size_t)bh * DD * SS;
  const unsigned char* Mb = mask + ((size_t)bh * SS + q0b) * SS;

  // cooperative staging assignments (256 threads, 16B each per tile)
  const int kr = tid >> 3, kg = tid & 7;  // K: 32 rows x 8 groups of 8 bf16
  const int vr = tid >> 2, vg = tid & 3;  // V: 64 rows x 4 groups of 8 bf16
  const int mr = tid >> 1, mg = tid & 1;  // M: 128 rows x 2 groups of 16 bytes

  auto stage = [&](int buf, int sk) {
    cp_async16(Kb + (size_t)(sk + kr) * DD + kg * 8, &Klds[buf][kr][kg * 8]);
    cp_async16(Vb + (size_t)vr * SS + sk + vg * 8,   &Vlds[buf][vr][vg * 8]);
    cp_async16(Mb + (size_t)mr * SS + sk + mg * 16,  &Mlds[buf][mr][mg * 16]);
  };

  float mstat[8], lstat[8];
  v8f oacc[4];
#pragma unroll
  for (int r = 0; r < 8; ++r) { mstat[r] = -1.0e30f; lstat[r] = 0.f; }
#pragma unroll
  for (int t = 0; t < 4; ++t) oacc[t] = vzero8();

  const int nblk = SS / 32;
  stage(0, 0);
  for (int i = 0; i < nblk; ++i) {
    const int cur = i & 1;
    if (i + 1 < nblk) { stage(cur ^ 1, (i + 1) * 32); cp_wait<3>(); }
    else              { cp_wait<0>(); }
    __syncthreads();  // buffer `cur` published by all waves

    // ---- scores: S = Q * K^T (two 16-col subtiles, 2 D-chunks) ----
    v8f sa[2];
#pragma unroll
    for (int t = 0; t < 2; ++t) {
      Frag bk0, bk1;
      bk0.q[0] = *(int4*)&Klds[cur][16 * t + mrow][8 * hb];
      bk0.q[1] = *(int4*)&Klds[cur][16 * t + mrow][16 + 8 * hb];
      bk1.q[0] = *(int4*)&Klds[cur][16 * t + mrow][32 + 8 * hb];
      bk1.q[1] = *(int4*)&Klds[cur][16 * t + mrow][48 + 8 * hb];
      sa[t] = wmma_bf16(aq[0], bk0, vzero8());
      sa[t] = wmma_bf16(aq[1], bk1, sa[t]);
    }

    // ---- mask (true -> -1e10); scores already scaled ----
    float sv[2][8];
    const int rbase = w * 16 + 8 * hb;
#pragma unroll
    for (int t = 0; t < 2; ++t) {
      const int col = 16 * t + mrow;
#pragma unroll
      for (int r = 0; r < 8; ++r) {
        const unsigned char mb = Mlds[cur][rbase + r][col];
        sv[t][r] = mb ? -1.0e10f : sa[t][r];
      }
    }

    // ---- online softmax (row stats within 16-lane halves) ----
    float alpha[8];
#pragma unroll
    for (int r = 0; r < 8; ++r) {
      float mx = fmaxf(sv[0][r], sv[1][r]);
#pragma unroll
      for (int off = 8; off >= 1; off >>= 1)
        mx = fmaxf(mx, __shfl_xor(mx, off, 32));
      const float mnew = fmaxf(mstat[r], mx);
      alpha[r] = __expf(mstat[r] - mnew);
      mstat[r] = mnew;
      const float p0 = __expf(sv[0][r] - mnew);
      const float p1 = __expf(sv[1][r] - mnew);
      sv[0][r] = p0;
      sv[1][r] = p1;
      float rs = p0 + p1;
#pragma unroll
      for (int off = 8; off >= 1; off >>= 1) rs += __shfl_xor(rs, off, 32);
      lstat[r] = lstat[r] * alpha[r] + rs;
    }
#pragma unroll
    for (int t = 0; t < 4; ++t)
#pragma unroll
      for (int r = 0; r < 8; ++r) oacc[t][r] *= alpha[r];

    // ---- P: C layout -> A layout via per-wave LDS tile ----
#pragma unroll
    for (int t = 0; t < 2; ++t)
#pragma unroll
      for (int r = 0; r < 8; ++r)
        Plds[w][r + 8 * hb][16 * t + mrow] = (__bf16)sv[t][r];
    asm volatile("" ::: "memory");  // fence; DS ops in-order per wave
    Frag pf;
    pf.q[0] = *(int4*)&Plds[w][mrow][8 * hb];
    pf.q[1] = *(int4*)&Plds[w][mrow][16 + 8 * hb];

    // ---- O += P * V (4 D-subtiles) ----
#pragma unroll
    for (int t = 0; t < 4; ++t) {
      Frag bv;
      bv.q[0] = *(int4*)&Vlds[cur][16 * t + mrow][8 * hb];
      bv.q[1] = *(int4*)&Vlds[cur][16 * t + mrow][16 + 8 * hb];
      oacc[t] = wmma_bf16(pf, bv, oacc[t]);
    }
    __syncthreads();  // all waves done reading `cur` before it is re-staged
  }

  // ---- normalize + store bf16 [B, S, H*D] ----
  const int b = bh >> 3, hh = bh & 7;
#pragma unroll
  for (int t = 0; t < 4; ++t)
#pragma unroll
    for (int r = 0; r < 8; ++r) {
      const int row = q0 + r + 8 * hb;
      const int d = 16 * t + mrow;
      const float v = oacc[t][r] / lstat[r];
      attnout[((size_t)b * SS + row) * HD + hh * DD + d] = (__bf16)v;
    }
}

// ---------------------------------------------------------------------------
extern "C" void kernel_launch(void* const* d_in, const int* in_sizes, int n_in,
                              void* d_out, int out_size, void* d_ws,
                              size_t ws_size, hipStream_t stream) {
  (void)in_sizes; (void)n_in; (void)out_size; (void)ws_size;
  const float* queries = (const float*)d_in[0];
  const float* keys    = (const float*)d_in[1];
  const float* values  = (const float*)d_in[2];
  const unsigned char* mask = (const unsigned char*)d_in[3];
  const float* Wq = (const float*)d_in[4];
  const float* Wk = (const float*)d_in[5];
  const float* Wv = (const float*)d_in[6];
  const float* Wo = (const float*)d_in[7];

  char* ws = (char*)d_ws;
  const size_t seg = (size_t)BB * HH * SS * DD * sizeof(__bf16);  // 8 MB
  __bf16* Qh = (__bf16*)(ws);
  __bf16* Kh = (__bf16*)(ws + seg);
  __bf16* Vt = (__bf16*)(ws + 2 * seg);
  __bf16* AO = (__bf16*)(ws + 3 * seg);

  dim3 gg(MM / 128, 512 / 64), bt(256);
  hipLaunchKernelGGL((gemm_wmma<0, 3>), gg, bt, 0, stream, (const void*)queries, Wq, (void*)Qh);
  hipLaunchKernelGGL((gemm_wmma<0, 1>), gg, bt, 0, stream, (const void*)keys,    Wk, (void*)Kh);
  hipLaunchKernelGGL((gemm_wmma<0, 2>), gg, bt, 0, stream, (const void*)values,  Wv, (void*)Vt);
  hipLaunchKernelGGL(flash_attn, dim3((BB * HH) * (SS / 128)), bt, 0, stream,
                     Qh, Kh, Vt, mask, AO);
  hipLaunchKernelGGL((gemm_wmma<1, 0>), gg, bt, 0, stream, (const void*)AO, Wo, d_out);
}
